// TransformerBlock_3435973837252
// MI455X (gfx1250) — matrix operand order repaired and checked by reading, not compiled, not run on hardware
//
#include <hip/hip_runtime.h>
#include <hip/hip_bf16.h>
#include <cstdint>

// ---------------- types ----------------
typedef __attribute__((ext_vector_type(16))) __bf16       v16bf;
typedef __attribute__((ext_vector_type(8)))  float        v8f;
typedef __attribute__((ext_vector_type(4)))  unsigned int u32x4;
typedef unsigned short u16;

constexpr int EMBED = 1024, HEADS = 16, HD = 64, FFN = 4096, SEQ = 2048, BATCH = 2;
constexpr int NTOK = BATCH * SEQ;  // 4096

// Async global->LDS path (gfx1250); guarded so either toolchain compiles.
#if defined(__HIP_DEVICE_COMPILE__) && __has_builtin(__builtin_amdgcn_global_load_async_to_lds_b128)
#define ASYNC_LDS 1
typedef int v4i_t __attribute__((vector_size(16)));
typedef __attribute__((address_space(1))) v4i_t* gv4i_p;   // global v4i32*
typedef __attribute__((address_space(3))) v4i_t* lv4i_p;   // LDS    v4i32*
#endif

union Frag {            // 32B = 8 VGPRs = one 16x32 bf16 WMMA operand
  v16bf v;
  u16   u[16];
  u32x4 q[2];
};

static __device__ __forceinline__ v8f vzero() {
  v8f z; for (int i = 0; i < 8; ++i) z[i] = 0.0f; return z;
}
static __device__ __forceinline__ v8f wmma_bf16(const Frag& a, const Frag& b, v8f c) {
  return __builtin_amdgcn_wmma_f32_16x16x32_bf16(false, a.v, false, b.v, (short)0, c, false, false);
}
static __device__ __forceinline__ u16 f2bf(float f) {  // round-to-nearest-even
  union { float f; unsigned u; } x; x.f = f;
  unsigned r = x.u + 0x7FFFu + ((x.u >> 16) & 1u);
  return (u16)(r >> 16);
}

// 32-byte global -> LDS stage: async on CDNA5 (ASYNCcnt), VGPR bounce otherwise.
static __device__ __forceinline__ void stage_cp32(u16* lds_dst, const u16* gsrc) {
#ifdef ASYNC_LDS
  __builtin_amdgcn_global_load_async_to_lds_b128(
      (gv4i_p)(unsigned long long)gsrc,
      (lv4i_p)(unsigned int)(unsigned long long)lds_dst, 0, 0);
  __builtin_amdgcn_global_load_async_to_lds_b128(
      (gv4i_p)(unsigned long long)(gsrc + 8),
      (lv4i_p)(unsigned int)(unsigned long long)(lds_dst + 8), 0, 0);
#else
  u32x4 q0 = ((const u32x4*)gsrc)[0];
  u32x4 q1 = ((const u32x4*)gsrc)[1];
  *(u32x4*)(lds_dst)     = q0;
  *(u32x4*)(lds_dst + 8) = q1;
#endif
}
static __device__ __forceinline__ void stage_wait() {
#ifdef ASYNC_LDS
#if __has_builtin(__builtin_amdgcn_s_wait_asynccnt)
  __builtin_amdgcn_s_wait_asynccnt(0);
#else
  asm volatile("s_wait_asynccnt 0" ::: "memory");
#endif
#endif
}

// ---------------- fp32 -> bf16 convert ----------------
__global__ void cvt_bf16_kernel(const float* __restrict__ in, u16* __restrict__ out, int n) {
  int i = blockIdx.x * blockDim.x + threadIdx.x;
  if (i < n) out[i] = f2bf(in[i]);
}

// ---------------- QKV projection: [NTOK*HEADS,64] x [64,64]^T (x3) ----------------
__global__ void qkv_kernel(const u16* __restrict__ xb,
                           const u16* __restrict__ wq, const u16* __restrict__ wk,
                           const u16* __restrict__ wv,
                           u16* __restrict__ qb, u16* __restrict__ kb, u16* __restrict__ vb) {
  const int lane = threadIdx.x & 31, wid = threadIdx.x >> 5;
  const int g = lane >> 4, l16 = lane & 15;
  const int rowbase = blockIdx.x * 128 + wid * 16;

  const u16* ar = xb + (size_t)(rowbase + l16) * 64;
  Frag a0, a1;
  a0.q[0] = *(const u32x4*)(ar +      16 * g);
  a0.q[1] = *(const u32x4*)(ar +      16 * g + 8);
  a1.q[0] = *(const u32x4*)(ar + 32 + 16 * g);
  a1.q[1] = *(const u32x4*)(ar + 32 + 16 * g + 8);

  v8f qa[4], ka[4], va[4];
  for (int t = 0; t < 4; ++t) { qa[t] = vzero(); ka[t] = vzero(); va[t] = vzero(); }

  for (int nt = 0; nt < 4; ++nt) {
    const int wrow = nt * 16 + l16;
    Frag b0, b1;
    b0.q[0] = *(const u32x4*)(wq + wrow * 64 + 16 * g);
    b0.q[1] = *(const u32x4*)(wq + wrow * 64 + 16 * g + 8);
    b1.q[0] = *(const u32x4*)(wq + wrow * 64 + 32 + 16 * g);
    b1.q[1] = *(const u32x4*)(wq + wrow * 64 + 32 + 16 * g + 8);
    qa[nt] = wmma_bf16(a0, b0, qa[nt]); qa[nt] = wmma_bf16(a1, b1, qa[nt]);
    b0.q[0] = *(const u32x4*)(wk + wrow * 64 + 16 * g);
    b0.q[1] = *(const u32x4*)(wk + wrow * 64 + 16 * g + 8);
    b1.q[0] = *(const u32x4*)(wk + wrow * 64 + 32 + 16 * g);
    b1.q[1] = *(const u32x4*)(wk + wrow * 64 + 32 + 16 * g + 8);
    ka[nt] = wmma_bf16(a0, b0, ka[nt]); ka[nt] = wmma_bf16(a1, b1, ka[nt]);
    b0.q[0] = *(const u32x4*)(wv + wrow * 64 + 16 * g);
    b0.q[1] = *(const u32x4*)(wv + wrow * 64 + 16 * g + 8);
    b1.q[0] = *(const u32x4*)(wv + wrow * 64 + 32 + 16 * g);
    b1.q[1] = *(const u32x4*)(wv + wrow * 64 + 32 + 16 * g + 8);
    va[nt] = wmma_bf16(a0, b0, va[nt]); va[nt] = wmma_bf16(a1, b1, va[nt]);
  }

  for (int nt = 0; nt < 4; ++nt) {
    for (int v = 0; v < 8; ++v) {
      const int d = nt * 16 + l16;
      const int r = rowbase + v + 8 * g;           // global row in [n*s*h]
      const int n = r >> 15;
      const int rem = r & 32767;
      const int s = rem >> 4, h = rem & 15;
      const size_t o = ((size_t)((n * 16 + h) * 2048 + s)) * 64 + d;
      qb[o] = f2bf(qa[nt][v] * 0.03125f);           // fold 1/sqrt(1024)
      kb[o] = f2bf(ka[nt][v]);
      vb[o] = f2bf(va[nt][v]);
    }
  }
}

// ---------------- flash attention: one wave = 16 query rows ----------------
__global__ void attn_kernel(const u16* __restrict__ qb, const u16* __restrict__ kb,
                            const u16* __restrict__ vb, u16* __restrict__ ctxb) {
  __shared__ u16   vT[64 * 40];
  __shared__ float Pbuf[4][16 * 32];

  const int lane = threadIdx.x & 31, wid = threadIdx.x >> 5;
  const int g = lane >> 4, l16 = lane & 15;
  const int b = blockIdx.x;
  const int qblk = b & 31, nh = b >> 5;
  const int h = nh & 15, n = nh >> 4;
  const int q0 = qblk * 64 + wid * 16;
  const size_t nhbase = (size_t)(n * 16 + h) * 2048;

  const u16* qr = qb + (nhbase + q0 + l16) * 64;
  Frag aq0, aq1;
  aq0.q[0] = *(const u32x4*)(qr +      16 * g);
  aq0.q[1] = *(const u32x4*)(qr +      16 * g + 8);
  aq1.q[0] = *(const u32x4*)(qr + 32 + 16 * g);
  aq1.q[1] = *(const u32x4*)(qr + 32 + 16 * g + 8);

  v8f acc[4]; for (int t = 0; t < 4; ++t) acc[t] = vzero();
  float m_run[8], l_run[8];
  for (int v = 0; v < 8; ++v) { m_run[v] = -1.0e30f; l_run[v] = 0.0f; }
  float* Pw = Pbuf[wid];

  for (int kt = 0; kt < SEQ; kt += 32) {
    {
      const int t = threadIdx.x;
      const int kl = t >> 2, d0 = (t & 3) * 16;
      const u16* src = vb + (nhbase + kt + kl) * 64 + d0;
      union { u32x4 q[2]; u16 u[16]; } uu;
      uu.q[0] = ((const u32x4*)src)[0];
      uu.q[1] = ((const u32x4*)src)[1];
      for (int i = 0; i < 16; ++i) vT[(d0 + i) * 40 + kl] = uu.u[i];
    }
    __syncthreads();

    Frag vbf[4];
    for (int dt = 0; dt < 4; ++dt) {
      const int base = (dt * 16 + l16) * 40 + 16 * g;
      vbf[dt].q[0] = *(const u32x4*)(vT + base);
      vbf[dt].q[1] = *(const u32x4*)(vT + base + 8);
    }

    v8f s[2];
    for (int st = 0; st < 2; ++st) {
      const u16* kr = kb + (nhbase + kt + st * 16 + l16) * 64;
      Frag bk0, bk1;
      bk0.q[0] = *(const u32x4*)(kr + 16 * g);
      bk0.q[1] = *(const u32x4*)(kr + 16 * g + 8);
      bk1.q[0] = *(const u32x4*)(kr + 32 + 16 * g);
      bk1.q[1] = *(const u32x4*)(kr + 32 + 16 * g + 8);
      v8f sv = wmma_bf16(aq0, bk0, vzero());
      s[st]  = wmma_bf16(aq1, bk1, sv);
    }

    for (int v = 0; v < 8; ++v) {
      float s0 = s[0][v], s1 = s[1][v];
      float mx = fmaxf(s0, s1);
      for (int off = 8; off > 0; off >>= 1) mx = fmaxf(mx, __shfl_xor(mx, off, 32));
      const float mnew = fmaxf(m_run[v], mx);
      const float al = __expf(m_run[v] - mnew);
      const float p0 = __expf(s0 - mnew), p1 = __expf(s1 - mnew);
      float rs = p0 + p1;
      for (int off = 8; off > 0; off >>= 1) rs += __shfl_xor(rs, off, 32);
      l_run[v] = l_run[v] * al + rs;
      m_run[v] = mnew;
      acc[0][v] *= al; acc[1][v] *= al; acc[2][v] *= al; acc[3][v] *= al;
      const int r = v + 8 * g;
      Pw[r * 32 + l16]      = p0;
      Pw[r * 32 + 16 + l16] = p1;
    }
    asm volatile("s_wait_dscnt 0" ::: "memory");  // within-wave LDS RAW (D-layout -> A-layout)

    Frag ap;
    for (int e = 0; e < 16; ++e) {
      const int kp = e + 16 * g;
      ap.u[e] = f2bf(Pw[l16 * 32 + kp]);
    }
    for (int dt = 0; dt < 4; ++dt) acc[dt] = wmma_bf16(ap, vbf[dt], acc[dt]);
    __syncthreads();
  }

  for (int dt = 0; dt < 4; ++dt) {
    for (int v = 0; v < 8; ++v) {
      const int d = dt * 16 + l16;
      const int r = v + 8 * g;
      const size_t token = (size_t)n * 2048 + q0 + r;
      ctxb[token * 1024 + h * 64 + d] = f2bf(acc[dt][v] / l_run[v]);
    }
  }
}

// ---------------- generic GEMM: C = act(A[M,K] @ W[N,K]^T + bias) ----------------
// Double-buffered LDS A-tiles filled with GLOBAL_LOAD_ASYNC_TO_LDS_B128 (ASYNCcnt),
// weights stream from the 192MB L2 directly into B fragments.
__global__ void gemm_kernel(const u16* __restrict__ A, const u16* __restrict__ W,
                            const float* __restrict__ bias,
                            float* __restrict__ outF, u16* __restrict__ outB,
                            int M, int N, int K, int relu) {
  __shared__ u16 As[2][128 * 40];

  const int lane = threadIdx.x & 31, wid = threadIdx.x >> 5;
  const int g = lane >> 4, l16 = lane & 15;
  const int nb = N >> 7;
  const int m0 = (blockIdx.x / nb) * 128, n0 = (blockIdx.x % nb) * 128;
  const int wm = wid & 1, wn = wid >> 1;   // 2 (M) x 4 (N) wave grid

  v8f acc[4][2];
  for (int i = 0; i < 4; ++i) for (int j = 0; j < 2; ++j) acc[i][j] = vzero();

  const int srow = threadIdx.x >> 1, sseg = (threadIdx.x & 1) * 16;
  const u16* arow = A + (size_t)(m0 + srow) * K + sseg;
  u16* const sdst0 = &As[0][srow * 40 + sseg];
  u16* const sdst1 = &As[1][srow * 40 + sseg];

  // prologue: fill buffer 0
  stage_cp32(sdst0, arow);
  stage_wait();
  __syncthreads();

  int buf = 0;
  for (int kt = 0; kt < K; kt += 32) {
    // overlap: kick off next K-tile into the other buffer while we do math
    if (kt + 32 < K) {
      stage_cp32(buf ? sdst0 : sdst1, arow + kt + 32);
      if (kt + 64 < K)
        __builtin_prefetch(arow + kt + 64, 0, 1);
    }

    Frag bf[2];
    for (int j = 0; j < 2; ++j) {
      const u16* wp = W + (size_t)(n0 + wn * 32 + j * 16 + l16) * K + kt + 16 * g;
      bf[j].q[0] = ((const u32x4*)wp)[0];
      bf[j].q[1] = ((const u32x4*)wp)[1];
    }
    const u16* Ab = As[buf];
    Frag af[4];
    for (int i = 0; i < 4; ++i) {
      const int base = (wm * 64 + i * 16 + l16) * 40;
      af[i].q[0] = *(const u32x4*)(Ab + base + 16 * g);
      af[i].q[1] = *(const u32x4*)(Ab + base + 16 * g + 8);
    }
    for (int i = 0; i < 4; ++i)
      for (int j = 0; j < 2; ++j)
        acc[i][j] = wmma_bf16(af[i], bf[j], acc[i][j]);

    stage_wait();       // next buffer's async copies landed
    __syncthreads();    // everyone done reading current buffer
    buf ^= 1;
  }

  for (int i = 0; i < 4; ++i) {
    for (int j = 0; j < 2; ++j) {
      const int nn = n0 + wn * 32 + j * 16 + l16;
      for (int v = 0; v < 8; ++v) {
        const int mm = m0 + wm * 64 + i * 16 + v + 8 * g;
        float val = acc[i][j][v] + bias[nn];
        if (relu) val = fmaxf(val, 0.0f);
        const size_t o = (size_t)mm * N + nn;
        if (outF) outF[o] = val;
        if (outB) outB[o] = f2bf(val);
      }
    }
  }
}

// ---------------- residual + LayerNorm (one block per 1024-wide row) ----------------
__global__ void ln_kernel(const float* __restrict__ a, const float* __restrict__ res,
                          const float* __restrict__ gamma, const float* __restrict__ beta,
                          float* __restrict__ outF, u16* __restrict__ outB) {
  __shared__ float red[256];
  const int row = blockIdx.x, t = threadIdx.x;
  const size_t base = (size_t)row * 1024;

  float y[4]; float s = 0.0f;
  for (int p = 0; p < 4; ++p) {
    const int c = t + p * 256;
    y[p] = a[base + c] + res[base + c];
    s += y[p];
  }
  red[t] = s; __syncthreads();
  for (int off = 128; off > 0; off >>= 1) { if (t < off) red[t] += red[t + off]; __syncthreads(); }
  const float mu = red[0] * (1.0f / 1024.0f); __syncthreads();

  s = 0.0f;
  for (int p = 0; p < 4; ++p) { const float d = y[p] - mu; s += d * d; }
  red[t] = s; __syncthreads();
  for (int off = 128; off > 0; off >>= 1) { if (t < off) red[t] += red[t + off]; __syncthreads(); }
  const float rstd = rsqrtf(red[0] * (1.0f / 1024.0f) + 1e-5f);

  for (int p = 0; p < 4; ++p) {
    const int c = t + p * 256;
    const float o = (y[p] - mu) * rstd * gamma[c] + beta[c];
    outF[base + c] = o;
    if (outB) outB[base + c] = f2bf(o);
  }
}

// ---------------- launcher ----------------
extern "C" void kernel_launch(void* const* d_in, const int* in_sizes, int n_in,
                              void* d_out, int out_size, void* d_ws, size_t ws_size,
                              hipStream_t stream) {
  const float* x   = (const float*)d_in[0];
  const float* Wq  = (const float*)d_in[1];
  const float* Wk  = (const float*)d_in[2];
  const float* Wv  = (const float*)d_in[3];
  const float* Wo  = (const float*)d_in[4];
  const float* bo  = (const float*)d_in[5];
  const float* g1  = (const float*)d_in[6];
  const float* be1 = (const float*)d_in[7];
  const float* g2  = (const float*)d_in[8];
  const float* be2 = (const float*)d_in[9];
  const float* W1  = (const float*)d_in[10];
  const float* b1  = (const float*)d_in[11];
  const float* W2  = (const float*)d_in[12];
  const float* b2  = (const float*)d_in[13];
  float* out = (float*)d_out;

  char* ws = (char*)d_ws;
  constexpr size_t B_XB  = (size_t)NTOK * EMBED * 2;
  constexpr size_t B_WSM = 64 * 64 * 2;
  constexpr size_t B_WOB = (size_t)EMBED * EMBED * 2;
  constexpr size_t B_W1B = (size_t)FFN * EMBED * 2;
  constexpr size_t B_QKV = (size_t)NTOK * EMBED * 2;
  constexpr size_t B_F32 = (size_t)NTOK * EMBED * 4;
  constexpr size_t B_HB  = (size_t)NTOK * FFN * 2;

  size_t o = 0;
  u16* xb  = (u16*)(ws + o); o += B_XB;
  u16* wqb = (u16*)(ws + o); o += B_WSM;
  u16* wkb = (u16*)(ws + o); o += B_WSM;
  u16* wvb = (u16*)(ws + o); o += B_WSM;
  u16* wob = (u16*)(ws + o); o += B_WOB;
  u16* w1b = (u16*)(ws + o); o += B_W1B;
  u16* w2b = (u16*)(ws + o); o += B_W1B;
  u16* qb  = (u16*)(ws + o); o += B_QKV;
  u16* kb  = (u16*)(ws + o); o += B_QKV;
  u16* vb  = (u16*)(ws + o); o += B_QKV;
  u16* ctxb  = (u16*)(ws + o); o += B_QKV;
  float* ao  = (float*)(ws + o); o += B_F32;
  float* x1  = (float*)(ws + o); o += B_F32;
  u16* x1b = (u16*)(ws + o); o += B_QKV;
  u16* hb  = (u16*)(ws + o); o += B_HB;
  float* ff  = (float*)(ws + o); o += B_F32;

  const int NE = NTOK * EMBED;          // 4194304
  cvt_bf16_kernel<<<(NE + 255) / 256, 256, 0, stream>>>(x,  xb,  NE);
  cvt_bf16_kernel<<<16, 256, 0, stream>>>(Wq, wqb, 4096);
  cvt_bf16_kernel<<<16, 256, 0, stream>>>(Wk, wkb, 4096);
  cvt_bf16_kernel<<<16, 256, 0, stream>>>(Wv, wvb, 4096);
  cvt_bf16_kernel<<<(EMBED * EMBED + 255) / 256, 256, 0, stream>>>(Wo, wob, EMBED * EMBED);
  cvt_bf16_kernel<<<(FFN * EMBED + 255) / 256, 256, 0, stream>>>(W1, w1b, FFN * EMBED);
  cvt_bf16_kernel<<<(FFN * EMBED + 255) / 256, 256, 0, stream>>>(W2, w2b, FFN * EMBED);

  qkv_kernel<<<(NTOK * HEADS) / 128, 256, 0, stream>>>(xb, wqb, wkb, wvb, qb, kb, vb);

  attn_kernel<<<BATCH * HEADS * (SEQ / 64), 128, 0, stream>>>(qb, kb, vb, ctxb);

  gemm_kernel<<<(NTOK / 128) * (EMBED / 128), 256, 0, stream>>>(
      ctxb, wob, bo, ao, (u16*)nullptr, NTOK, EMBED, EMBED, 0);
  ln_kernel<<<NTOK, 256, 0, stream>>>(ao, x, g1, be1, x1, x1b);
  gemm_kernel<<<(NTOK / 128) * (FFN / 128), 256, 0, stream>>>(
      x1b, w1b, b1, (float*)nullptr, hb, NTOK, FFN, EMBED, 1);
  gemm_kernel<<<(NTOK / 128) * (EMBED / 128), 256, 0, stream>>>(
      hb, w2b, b2, ff, (u16*)nullptr, NTOK, EMBED, FFN, 0);
  ln_kernel<<<NTOK, 256, 0, stream>>>(ff, x1, g2, be2, out, (u16*)nullptr);
}
